// Model_Geom_Graph_34153579938554
// MI455X (gfx1250) — matrix operand order repaired and checked by reading, not compiled, not run on hardware
//
#include <hip/hip_runtime.h>
#include <hip/hip_bf16.h>
#include <hip/hip_fp16.h>

// ---------------------------------------------------------------------------
// GATv2 GNN forward for MI455X (gfx1250, wave32, WMMA).
// Node-transform GEMMs use v_wmma_f32_16x16x32_f16; edge phase uses mapped
// atomic-max (segment max), atomic-add (segment sum) — L2-resident (192MB L2).
// xl/xr are padded to NPAD rows so the WMMA store epilogue needs no guards.
// ---------------------------------------------------------------------------

#define NN    50000   // nodes
#define NPAD  50016   // nodes padded to 16
#define NE    600000  // real edges
#define NET   650000  // edges + self loops
#define NB    50      // batch segments
#define HC    128

typedef __attribute__((ext_vector_type(16))) _Float16 v16h;
typedef __attribute__((ext_vector_type(8)))  float    v8f;

union HFrag { uint4 u[2]; v16h h; };

__device__ __forceinline__ unsigned enc_f32(float f) {
  unsigned u = __float_as_uint(f);
  return (u >> 31) ? ~u : (u | 0x80000000u);   // monotone map for unsigned max
}
__device__ __forceinline__ float dec_f32(unsigned u) {
  unsigned v = (u >> 31) ? (u & 0x7fffffffu) : ~u;
  return __uint_as_float(v);
}

// ---------------- WMMA GEMM: C[NPAD x 128] = A[NPAD x K](f16) * W[K x 128](f16) + bias
// block = 256 threads (8 waves); wave w owns column tile w (16 cols);
// block row = 16 rows. A-frag / B-frag layouts per CDNA5 ISA 7.12.2.
// Stores are unguarded: C has NPAD rows; pad rows are never consumed.
template <int K>
__global__ __launch_bounds__(256)
void gemm_node(const _Float16* __restrict__ A, const _Float16* __restrict__ W,
               const float* __restrict__ bias, float* __restrict__ C) {
  const int lane = threadIdx.x & 31;
  const int wave = threadIdx.x >> 5;
  const int row0 = blockIdx.x << 4;
  const int nb   = wave << 4;
  const int r    = row0 + (lane & 15);
  const int kb   = (lane < 16) ? 0 : 8;
  v8f acc = {};
#pragma unroll
  for (int kc = 0; kc < K; kc += 32) {
    HFrag a, b;
    // A 16x32 f16: lane holds row r, halfs K = kb..kb+7 and kb+16..kb+23
    const _Float16* ap = A + (size_t)r * K + (kc + kb);
    a.u[0] = *(const uint4*)(ap);
    a.u[1] = *(const uint4*)(ap + 16);
    // B 32x16 f16: lane holds col N = nb+(lane&15), halfs K = kb..kb+7 and kb+16..kb+23
    const _Float16* bp = W + (size_t)(kc + kb) * HC + nb + (lane & 15);
#pragma unroll
    for (int s = 0; s < 8; ++s) {
      b.h[s]     = bp[(size_t)s * HC];
      b.h[8 + s] = bp[(size_t)(16 + s) * HC];
    }
    acc = __builtin_amdgcn_wmma_f32_16x16x32_f16(false, a.h, false, b.h,
                                                 (short)0, acc, false, false);
  }
  // C 16x16 f32: vgpr i -> M = row0 + i (+8 for lanes 16..31), N = nb + (lane&15)
  const int n  = nb + (lane & 15);
  const float bv = bias[n];
  const int mb = row0 + ((lane < 16) ? 0 : 8);
  float* cp = C + (size_t)mb * HC + n;
#pragma unroll
  for (int i = 0; i < 8; ++i) cp[(size_t)i * HC] = acc[i] + bv;  // imm-offset stores
}

// ---------------- helpers --------------------------------------------------
__global__ void zero32(unsigned* __restrict__ p, long long n) {
  long long t = (long long)blockIdx.x * 256 + threadIdx.x;
  if (t < n) p[t] = 0u;
}

__global__ void cvt_f2h(const float* __restrict__ s, _Float16* __restrict__ d, int n) {
  int t = blockIdx.x * 256 + threadIdx.x;
  if (t < n) d[t] = (_Float16)s[t];
}

// pad nodes [NN x 4] -> f16 [NPAD x 32] (zeros elsewhere)
__global__ void pad_nodes(const float* __restrict__ x, _Float16* __restrict__ xh) {
  int t = blockIdx.x * 256 + threadIdx.x;
  if (t >= NPAD * 32) return;
  int n = t >> 5, k = t & 31;
  float v = (n < NN && k < 4) ? x[n * 4 + k] : 0.0f;
  xh[t] = (_Float16)v;
}

// pad W0 [4 x 128] -> f16 [32 x 128] (zeros elsewhere)
__global__ void pad_w0(const float* __restrict__ w, _Float16* __restrict__ wh) {
  int t = blockIdx.x * 256 + threadIdx.x;
  if (t >= 32 * HC) return;
  int k = t >> 7;
  wh[t] = (_Float16)((k < 4) ? w[t] : 0.0f);
}

// ---------------- edge phase ----------------------------------------------
// wave per edge: z = lrelu(xl[s]+xr[d]+ea@We, 0.2); logit[h] = sum_c z*att
__global__ __launch_bounds__(256)
void edge_logit(const int* __restrict__ ei, const float* __restrict__ ea,
                const float* __restrict__ xl, const float* __restrict__ xr,
                const float* __restrict__ We, const float* __restrict__ att,
                float* __restrict__ logit, unsigned* __restrict__ mx) {
  int e = blockIdx.x * 8 + (threadIdx.x >> 5);
  int lane = threadIdx.x & 31;
  if (e >= NET) return;
  int s, d; float e0, e1;
  if (e < NE) { s = ei[e]; d = ei[NE + e]; e0 = ea[2 * e]; e1 = ea[2 * e + 1]; }
  else        { s = d = e - NE; e0 = 0.f; e1 = 0.f; }
  for (int h = 0; h < 4; ++h) {
    int c = h * 32 + lane;
    float v = xl[(size_t)s * HC + c] + xr[(size_t)d * HC + c]
            + e0 * We[c] + e1 * We[HC + c];
    v = (v > 0.f) ? v : 0.2f * v;
    float p = v * att[c];
    for (int off = 16; off > 0; off >>= 1) p += __shfl_xor(p, off, 32);
    if (lane == 0) {
      logit[(size_t)e * 4 + h] = p;
      atomicMax(&mx[(size_t)d * 4 + h], enc_f32(p));
    }
  }
}

// ex = exp(logit - mx[dst]) (in place), den[dst] += ex
__global__ void edge_exp(const int* __restrict__ ei, float* __restrict__ lg,
                         const unsigned* __restrict__ mx, float* __restrict__ den) {
  long long t = (long long)blockIdx.x * 256 + threadIdx.x;
  if (t >= (long long)NET * 4) return;
  int e = (int)(t >> 2), h = (int)(t & 3);
  int d = (e < NE) ? ei[NE + e] : (e - NE);
  float m = dec_f32(mx[(size_t)d * 4 + h]);
  if (!__builtin_isfinite(m)) m = 0.f;
  float x = expf(lg[t] - m);
  lg[t] = x;
  atomicAdd(&den[(size_t)d * 4 + h], x);
}

// out[dst] += xl[src] * (ex / (den[dst] + 1e-16))
__global__ void scatter_agg(const int* __restrict__ ei, const float* __restrict__ xl,
                            const float* __restrict__ ex, const float* __restrict__ den,
                            float* __restrict__ outacc) {
  long long t = (long long)blockIdx.x * 256 + threadIdx.x;
  if (t >= (long long)NET * HC) return;
  int e = (int)(t >> 7), c = (int)(t & 127), h = c >> 5;
  int s, d;
  if (e < NE) { s = ei[e]; d = ei[NE + e]; } else { s = d = e - NE; }
  float alpha = ex[(size_t)e * 4 + h] / (den[(size_t)d * 4 + h] + 1e-16f);
  atomicAdd(&outacc[(size_t)d * HC + c], xl[(size_t)s * HC + c] * alpha);
}

// x = relu(outacc + bias) [+res]; optionally res = x; write f16 copy (pad rows = 0)
__global__ void finalize(const float* __restrict__ outacc, const float* __restrict__ bias,
                         float* __restrict__ x, float* __restrict__ res,
                         _Float16* __restrict__ xh, int add_res, int save_res) {
  int t = blockIdx.x * 256 + threadIdx.x;
  if (t >= NPAD * HC) return;
  int n = t >> 7, c = t & 127;
  if (n < NN) {
    float v = outacc[t] + bias[c];
    v = v > 0.f ? v : 0.f;
    if (add_res)  v += res[t];
    if (save_res) res[t] = v;
    x[t]  = v;
    xh[t] = (_Float16)v;
  } else {
    xh[t] = (_Float16)0.f;
  }
}

// ---------------- pooling --------------------------------------------------
__global__ void pool_scatter(const float* __restrict__ x, const int* __restrict__ batch,
                             float* __restrict__ sum, unsigned* __restrict__ maxe,
                             float* __restrict__ cnt) {
  int t = blockIdx.x * 256 + threadIdx.x;
  if (t >= NN * HC) return;
  int n = t >> 7, c = t & 127;
  int b = batch[n];
  float v = x[t];
  atomicAdd(&sum[b * HC + c], v);
  atomicMax(&maxe[b * HC + c], enc_f32(v));
  if (c == 0) atomicAdd(&cnt[b], 1.0f);
}

__global__ void pool_build(const float* __restrict__ sum, const unsigned* __restrict__ maxe,
                           const float* __restrict__ cnt, float* __restrict__ X) {
  int t = blockIdx.x * 256 + threadIdx.x;
  if (t >= NB * 256) return;
  int b = t >> 8, c = t & 255;
  float v;
  if (c < HC) {
    v = sum[b * HC + c] / fmaxf(cnt[b], 1.0f);
  } else {
    float m = dec_f32(maxe[b * HC + (c - HC)]);
    v = __builtin_isfinite(m) ? m : 0.f;
  }
  X[t] = v;
}

// ---------------- tiny dense trunk (50 rows, fp32) -------------------------
// act: 0=none, 1=relu, 2=tanh ; C[m*ldc + coff + n]
__global__ void dense(const float* __restrict__ A, const float* __restrict__ W,
                      const float* __restrict__ bias, float* __restrict__ C,
                      int M, int K, int N, int ldc, int coff, int act) {
  int t = blockIdx.x * 256 + threadIdx.x;
  if (t >= M * N) return;
  int m = t / N, n = t % N;
  float acc = bias[n];
  for (int k = 0; k < K; ++k) acc += A[m * K + k] * W[k * N + n];
  if (act == 1)      acc = fmaxf(acc, 0.f);
  else if (act == 2) acc = tanhf(acc);
  C[m * ldc + coff + n] = acc;
}

// ---------------------------------------------------------------------------
static inline unsigned gdiv(long long n) { return (unsigned)((n + 255) / 256); }

extern "C" void kernel_launch(void* const* d_in, const int* in_sizes, int n_in,
                              void* d_out, int out_size, void* d_ws, size_t ws_size,
                              hipStream_t stream) {
  (void)in_sizes; (void)n_in; (void)out_size; (void)ws_size;
  const float* nodes  = (const float*)d_in[0];
  const int*   ei     = (const int*)  d_in[1];
  const float* ea     = (const float*)d_in[2];
  const int*   batch  = (const int*)  d_in[3];
  const float* W0l    = (const float*)d_in[4];
  const float* b0l    = (const float*)d_in[5];
  const float* W0r    = (const float*)d_in[6];
  const float* b0r    = (const float*)d_in[7];
  const float* W0e    = (const float*)d_in[8];
  const float* att0   = (const float*)d_in[9];
  const float* bias0  = (const float*)d_in[10];
  const float* Wl     = (const float*)d_in[11];
  const float* bl     = (const float*)d_in[12];
  const float* Wr     = (const float*)d_in[13];
  const float* br     = (const float*)d_in[14];
  const float* Wee    = (const float*)d_in[15];
  const float* atts   = (const float*)d_in[16];
  const float* biases = (const float*)d_in[17];
  const float* D1_W   = (const float*)d_in[18];
  const float* D1_b   = (const float*)d_in[19];
  const float* D23_W  = (const float*)d_in[20];
  const float* D23_b  = (const float*)d_in[21];
  const float* H1_W   = (const float*)d_in[22];
  const float* H1_b   = (const float*)d_in[23];
  const float* H2_W   = (const float*)d_in[24];
  const float* H2_b   = (const float*)d_in[25];
  const float* H3_W   = (const float*)d_in[26];
  const float* H3_b   = (const float*)d_in[27];
  float* out = (float*)d_out;

  // ---- carve workspace (~160 MB) ----
  char* p = (char*)d_ws;
  auto carve = [&](size_t bytes) { void* r = (void*)p; p += (bytes + 255) & ~(size_t)255; return r; };
  _Float16* xh   = (_Float16*)carve((size_t)NPAD * HC * 2);
  _Float16* xh0  = (_Float16*)carve((size_t)NPAD * 32 * 2);
  _Float16* Whl  = (_Float16*)carve((size_t)8 * HC * HC * 2);
  _Float16* Whr  = (_Float16*)carve((size_t)8 * HC * HC * 2);
  _Float16* W0lh = (_Float16*)carve((size_t)32 * HC * 2);
  _Float16* W0rh = (_Float16*)carve((size_t)32 * HC * 2);
  float*    xf   = (float*)   carve((size_t)NN * HC * 4);
  float*    res  = (float*)   carve((size_t)NN * HC * 4);
  float*    xl   = (float*)   carve((size_t)NPAD * HC * 4);  // padded: unguarded stores
  float*    xr   = (float*)   carve((size_t)NPAD * HC * 4);  // padded: unguarded stores
  float*    oac  = (float*)   carve((size_t)NN * HC * 4);
  float*    lg   = (float*)   carve((size_t)NET * 4 * 4);    // logits -> ex (in place)
  unsigned* mx   = (unsigned*)carve((size_t)NN * 4 * 4);
  float*    den  = (float*)   carve((size_t)NN * 4 * 4);
  float*    psum = (float*)   carve((size_t)NB * HC * 4);
  unsigned* pmax = (unsigned*)carve((size_t)NB * HC * 4);
  float*    pcnt = (float*)   carve((size_t)NB * 4);
  float*    X    = (float*)   carve((size_t)NB * 256 * 4);
  float*    T1   = (float*)   carve((size_t)NB * HC * 4);
  float*    T2   = (float*)   carve((size_t)NB * HC * 4);
  float*    T3   = (float*)   carve((size_t)NB * HC * 4);
  float*    Y1   = (float*)   carve((size_t)NB * HC * 4);
  float*    Y2   = (float*)   carve((size_t)NB * 64 * 4);

  // ---- prep: convert weights / inputs to f16 ----
  cvt_f2h<<<gdiv(8 * HC * HC), 256, 0, stream>>>(Wl, Whl, 8 * HC * HC);
  cvt_f2h<<<gdiv(8 * HC * HC), 256, 0, stream>>>(Wr, Whr, 8 * HC * HC);
  pad_w0<<<gdiv(32 * HC), 256, 0, stream>>>(W0l, W0lh);
  pad_w0<<<gdiv(32 * HC), 256, 0, stream>>>(W0r, W0rh);
  pad_nodes<<<gdiv((long long)NPAD * 32), 256, 0, stream>>>(nodes, xh0);

  // ---- one GATv2 layer ----
  auto layer = [&](const _Float16* Ah, int K, const _Float16* Wlh, const float* blv,
                   const _Float16* Wrh, const float* brv, const float* Wev,
                   const float* attv, const float* biasv, int add_res, int save_res) {
    zero32<<<gdiv((long long)NN * HC), 256, 0, stream>>>((unsigned*)oac, (long long)NN * HC);
    zero32<<<gdiv((long long)NN * 4), 256, 0, stream>>>(mx, (long long)NN * 4);
    zero32<<<gdiv((long long)NN * 4), 256, 0, stream>>>((unsigned*)den, (long long)NN * 4);
    if (K == 32) {
      gemm_node<32><<<NPAD / 16, 256, 0, stream>>>(Ah, Wlh, blv, xl);
      gemm_node<32><<<NPAD / 16, 256, 0, stream>>>(Ah, Wrh, brv, xr);
    } else {
      gemm_node<128><<<NPAD / 16, 256, 0, stream>>>(Ah, Wlh, blv, xl);
      gemm_node<128><<<NPAD / 16, 256, 0, stream>>>(Ah, Wrh, brv, xr);
    }
    edge_logit<<<gdiv((long long)NET * 32), 256, 0, stream>>>(ei, ea, xl, xr, Wev, attv, lg, mx);
    edge_exp<<<gdiv((long long)NET * 4), 256, 0, stream>>>(ei, lg, mx, den);
    scatter_agg<<<gdiv((long long)NET * HC), 256, 0, stream>>>(ei, xl, lg, den, oac);
    finalize<<<gdiv((long long)NPAD * HC), 256, 0, stream>>>(oac, biasv, xf, res, xh, add_res, save_res);
  };

  // conv0 (saves residual for block 0)
  layer(xh0, 32, W0lh, b0l, W0rh, b0r, W0e, att0, bias0, /*add*/0, /*save*/1);
  // 8 stacked layers; residual added after each odd layer, then re-saved
  for (int j = 0; j < 8; ++j) {
    layer(xh, 128, Whl + (size_t)j * HC * HC, bl + (size_t)j * HC,
          Whr + (size_t)j * HC * HC, br + (size_t)j * HC,
          Wee + (size_t)j * 2 * HC, atts + (size_t)j * 4 * 32,
          biases + (size_t)j * HC, /*add*/ (j & 1), /*save*/ (j & 1));
  }

  // ---- pooling: mean + max per batch segment ----
  zero32<<<gdiv(NB * HC), 256, 0, stream>>>((unsigned*)psum, NB * HC);
  zero32<<<gdiv(NB * HC), 256, 0, stream>>>(pmax, NB * HC);
  zero32<<<1, 256, 0, stream>>>((unsigned*)pcnt, NB);
  pool_scatter<<<gdiv((long long)NN * HC), 256, 0, stream>>>(xf, batch, psum, pmax, pcnt);
  pool_build<<<gdiv(NB * 256), 256, 0, stream>>>(psum, pmax, pcnt, X);

  // ---- dense trunk ----
  dense<<<gdiv(NB * HC), 256, 0, stream>>>(X,  D1_W,            D1_b,        T1, NB, 256, HC, HC, 0, 1);
  dense<<<gdiv(NB * HC), 256, 0, stream>>>(T1, D23_W,           D23_b,       T2, NB, HC, HC, HC, 0, 1);
  dense<<<gdiv(NB * HC), 256, 0, stream>>>(T2, D23_W + HC * HC, D23_b + HC,  T3, NB, HC, HC, HC, 0, 1);

  // ---- 3 output heads: Chi0 (tanh), Rp, T0 ----
  for (int h = 0; h < 3; ++h) {
    dense<<<gdiv(NB * HC), 256, 0, stream>>>(T3, H1_W + (size_t)h * HC * HC, H1_b + h * HC,
                                             Y1, NB, HC, HC, HC, 0, 1);
    dense<<<gdiv(NB * 64), 256, 0, stream>>>(Y1, H2_W + (size_t)h * HC * 64, H2_b + h * 64,
                                             Y2, NB, HC, 64, 64, 0, 1);
    dense<<<1, 256, 0, stream>>>(Y2, H3_W + (size_t)h * 64, H3_b + h,
                                 out, NB, 64, 1, 3, h, (h == 0) ? 2 : 0);
  }
}
